// EncLayer_38208029065286
// MI455X (gfx1250) — compile-verified, matrix operand order checked by reading, not run on hardware
//
#include <hip/hip_runtime.h>
#include <hip/hip_bf16.h>

typedef __bf16 bf16_t;
typedef __attribute__((ext_vector_type(16))) __bf16 v16bf;
typedef __attribute__((ext_vector_type(8)))  float  v8f;

#define B_DIM  4
#define N_DIM  2048
#define K_NB   48
#define H_DIM  128
#define C_IN   256      // H + EDGE
#define FF_DIM 512
#define ROWS2  32       // rows per FFN block

union V16 {
  v16bf v;
  unsigned u[8];
  uint4 q[2];
};

// ---- WMMA operand loaders -------------------------------------------------
// A-operand (16-bit, 16x32): lane = M%16 + 16*half; per-lane 16 bf16 cover
// K in {0..7,16..23} (+8 if half) packed 2-per-VGPR per CDNA5 ISA.
__device__ __forceinline__ v16bf load_A_tile(const bf16_t* base, int rs,
                                             int rowBase, int kBase, int lane) {
  const int m  = rowBase + (lane & 15);
  const int hl = (lane >> 4) & 1;
  const bf16_t* row = base + m * rs + kBase + 8 * hl;
  V16 r;
#pragma unroll
  for (int v = 0; v < 8; ++v) {
    const int off = ((v & 4) << 2) + 2 * (v & 3);   // 0,2,4,6,16,18,20,22
    r.u[v] = *(const unsigned*)(row + off);         // two bf16 per 32-bit read
  }
  return r.v;
}

// B-operand: pre-packed in global ws as [tile][lane][16] bf16 -> 2x b128 loads.
__device__ __forceinline__ v16bf load_B_tile(const bf16_t* __restrict__ wpk,
                                             int tile, int lane) {
  const uint4* p = (const uint4*)(wpk + (((size_t)tile * 32 + lane) << 4));
  V16 r;
  r.q[0] = p[0];
  r.q[1] = p[1];
  return r.v;
}

__device__ __forceinline__ v8f wmma_bf16(v16bf a, v16bf b, v8f c) {
  return __builtin_amdgcn_wmma_f32_16x16x32_bf16(false, a, false, b,
                                                 (short)0, c, false, false);
}

__device__ __forceinline__ float gelu_tanh_f(float x) {
  const float k0 = 0.7978845608028654f;
  const float k1 = 0.044715f;
  return 0.5f * x * (1.0f + tanhf(k0 * (x + k1 * x * x * x)));
}
__device__ __forceinline__ float gelu_erf_f(float x) {
  return 0.5f * x * (1.0f + erff(x * 0.70710678118654752440f));
}

// ---- Weight pre-pack: fp32 row-major [Kc x N] -> bf16 WMMA B-tile layout ---
__global__ void pack_wmma_b(const float* __restrict__ W, bf16_t* __restrict__ Wpk,
                            int Kc, int N) {
  int i = blockIdx.x * blockDim.x + threadIdx.x;
  if (i >= Kc * N) return;
  int e    = i & 15;
  int lane = (i >> 4) & 31;
  int tile = i >> 9;
  int ntiles = N >> 4;
  int tk = tile / ntiles;
  int tn = tile - tk * ntiles;
  int v = e >> 1, rbit = e & 1, hl = lane >> 4, n = lane & 15;
  int k = ((v & 4) << 2) + 2 * (v & 3) + 8 * hl + rbit;
  Wpk[i] = (bf16_t)W[(size_t)(tk * 32 + k) * N + tn * 16 + n];
}

// ---- Kernel 1: gather + 3-layer message MLP + masked K-reduction -----------
__global__ __launch_bounds__(128)
void enc_msg_kernel(const float* __restrict__ h_V, const float* __restrict__ h_E,
                    const int* __restrict__ E_idx, const float* __restrict__ mask_attend,
                    const bf16_t* __restrict__ W1pk, const float* __restrict__ W1_b,
                    const bf16_t* __restrict__ W2pk, const float* __restrict__ W2_b,
                    const bf16_t* __restrict__ W3pk, const float* __restrict__ W3_b,
                    float* __restrict__ hbuf) {
  constexpr int RSX = C_IN + 8;    // 264 (padded, breaks LDS bank conflicts)
  constexpr int RSM = H_DIM + 8;   // 136
  __shared__ bf16_t X[K_NB * RSX];    // h_EV, later reused for m2
  __shared__ bf16_t M1[K_NB * RSM];   // m1
  __shared__ float  colsum[H_DIM];
  __shared__ int    idx_s[K_NB];
  __shared__ float  mask_s[K_NB];

  const int tid  = threadIdx.x;
  const int lane = tid & 31;
  const int wave = tid >> 5;
  const int bn   = blockIdx.x;            // 0 .. B*N-1
  const int b    = bn / N_DIM;

  if (tid < K_NB) {
    idx_s[tid]  = E_idx[(size_t)bn * K_NB + tid];
    mask_s[tid] = mask_attend[(size_t)bn * K_NB + tid];
  }
  if (tid < H_DIM) colsum[tid] = 0.0f;
  __syncthreads();

  // Build h_EV = [gather(h_V, idx) | h_E] as bf16 in LDS (48 x 256).
  const float* hv_b = h_V + (size_t)b * N_DIM * H_DIM;
  const float* he_b = h_E + (size_t)bn * K_NB * H_DIM;   // EDGE == 128
  for (int r = 0; r < K_NB; ++r) {
    X[r * RSX + tid]         = (bf16_t)hv_b[(size_t)idx_s[r] * H_DIM + tid];
    X[r * RSX + H_DIM + tid] = (bf16_t)he_b[(size_t)r * H_DIM + tid];
  }
  __syncthreads();

  const int hl = lane >> 4;
  const v8f vzero = {0.f, 0.f, 0.f, 0.f, 0.f, 0.f, 0.f, 0.f};
  v8f acc[3][2];

  // ---- GEMM1: m1 = gelu_tanh(X[48x256] @ W1[256x128] + b1) ----
#pragma unroll
  for (int rt = 0; rt < 3; ++rt)
#pragma unroll
    for (int j = 0; j < 2; ++j) acc[rt][j] = vzero;
#pragma unroll
  for (int ks = 0; ks < 8; ++ks) {
    v16bf a0 = load_A_tile(X, RSX, 0, ks * 32, lane);
    v16bf a1 = load_A_tile(X, RSX, 16, ks * 32, lane);
    v16bf a2 = load_A_tile(X, RSX, 32, ks * 32, lane);
#pragma unroll
    for (int j = 0; j < 2; ++j) {
      v16bf bt = load_B_tile(W1pk, ks * 8 + (wave * 2 + j), lane);
      acc[0][j] = wmma_bf16(a0, bt, acc[0][j]);
      acc[1][j] = wmma_bf16(a1, bt, acc[1][j]);
      acc[2][j] = wmma_bf16(a2, bt, acc[2][j]);
    }
  }
#pragma unroll
  for (int rt = 0; rt < 3; ++rt)
#pragma unroll
    for (int j = 0; j < 2; ++j) {
      int col = (wave * 2 + j) * 16 + (lane & 15);
      float bias = W1_b[col];
#pragma unroll
      for (int v = 0; v < 8; ++v) {
        int row = rt * 16 + v + 8 * hl;           // C/D layout: M = v + 8*half
        M1[row * RSM + col] = (bf16_t)gelu_tanh_f(acc[rt][j][v] + bias);
      }
    }
  __syncthreads();

  // ---- GEMM2: m2 = gelu_tanh(m1[48x128] @ W2[128x128] + b2) -> into X ----
#pragma unroll
  for (int rt = 0; rt < 3; ++rt)
#pragma unroll
    for (int j = 0; j < 2; ++j) acc[rt][j] = vzero;
#pragma unroll
  for (int ks = 0; ks < 4; ++ks) {
    v16bf a0 = load_A_tile(M1, RSM, 0, ks * 32, lane);
    v16bf a1 = load_A_tile(M1, RSM, 16, ks * 32, lane);
    v16bf a2 = load_A_tile(M1, RSM, 32, ks * 32, lane);
#pragma unroll
    for (int j = 0; j < 2; ++j) {
      v16bf bt = load_B_tile(W2pk, ks * 8 + (wave * 2 + j), lane);
      acc[0][j] = wmma_bf16(a0, bt, acc[0][j]);
      acc[1][j] = wmma_bf16(a1, bt, acc[1][j]);
      acc[2][j] = wmma_bf16(a2, bt, acc[2][j]);
    }
  }
#pragma unroll
  for (int rt = 0; rt < 3; ++rt)
#pragma unroll
    for (int j = 0; j < 2; ++j) {
      int col = (wave * 2 + j) * 16 + (lane & 15);
      float bias = W2_b[col];
#pragma unroll
      for (int v = 0; v < 8; ++v) {
        int row = rt * 16 + v + 8 * hl;
        X[row * RSX + col] = (bf16_t)gelu_tanh_f(acc[rt][j][v] + bias);
      }
    }
  __syncthreads();

  // ---- GEMM3: m3 = m2[48x128] @ W3[128x128] + b3; masked colsum over K ----
#pragma unroll
  for (int rt = 0; rt < 3; ++rt)
#pragma unroll
    for (int j = 0; j < 2; ++j) acc[rt][j] = vzero;
#pragma unroll
  for (int ks = 0; ks < 4; ++ks) {
    v16bf a0 = load_A_tile(X, RSX, 0, ks * 32, lane);
    v16bf a1 = load_A_tile(X, RSX, 16, ks * 32, lane);
    v16bf a2 = load_A_tile(X, RSX, 32, ks * 32, lane);
#pragma unroll
    for (int j = 0; j < 2; ++j) {
      v16bf bt = load_B_tile(W3pk, ks * 8 + (wave * 2 + j), lane);
      acc[0][j] = wmma_bf16(a0, bt, acc[0][j]);
      acc[1][j] = wmma_bf16(a1, bt, acc[1][j]);
      acc[2][j] = wmma_bf16(a2, bt, acc[2][j]);
    }
  }
#pragma unroll
  for (int rt = 0; rt < 3; ++rt)
#pragma unroll
    for (int j = 0; j < 2; ++j) {
      int col = (wave * 2 + j) * 16 + (lane & 15);
      float bias = W3_b[col];
      float part = 0.0f;
#pragma unroll
      for (int v = 0; v < 8; ++v) {
        int row = rt * 16 + v + 8 * hl;
        part += mask_s[row] * (acc[rt][j][v] + bias);
      }
      atomicAdd(&colsum[col], part);   // ds_add_f32
    }
  __syncthreads();

  if (tid < H_DIM) {
    float h = h_V[(size_t)bn * H_DIM + tid] + colsum[tid] * (1.0f / 30.0f);
    hbuf[(size_t)bn * H_DIM + tid] = h;
  }
}

// ---- Kernel 2: position-wise FFN + residual + mask_V -----------------------
__global__ __launch_bounds__(128)
void enc_ffn_kernel(const float* __restrict__ hbuf, const float* __restrict__ mask_V,
                    const bf16_t* __restrict__ Winpk, const float* __restrict__ Win_b,
                    const bf16_t* __restrict__ Woutpk, const float* __restrict__ Wout_b,
                    float* __restrict__ out) {
  constexpr int RSX = H_DIM + 8;    // 136
  constexpr int RSF = FF_DIM + 8;   // 520
  __shared__ bf16_t Xb[ROWS2 * RSX];
  __shared__ bf16_t Fb[ROWS2 * RSF];

  const int tid  = threadIdx.x;
  const int lane = tid & 31;
  const int wave = tid >> 5;
  const int hl   = lane >> 4;
  const int base = blockIdx.x * ROWS2;

  for (int r = 0; r < ROWS2; ++r)
    Xb[r * RSX + tid] = (bf16_t)hbuf[(size_t)(base + r) * H_DIM + tid];
  __syncthreads();

  const v8f vzero = {0.f, 0.f, 0.f, 0.f, 0.f, 0.f, 0.f, 0.f};

  // ff = gelu_exact(h[32x128] @ Win[128x512] + b); FF=512 -> 32 ntiles, 8/wave
  for (int nj = 0; nj < 8; ++nj) {
    const int tn = wave * 8 + nj;
    v8f a0c = vzero, a1c = vzero;
#pragma unroll
    for (int ks = 0; ks < 4; ++ks) {
      v16bf a0 = load_A_tile(Xb, RSX, 0, ks * 32, lane);
      v16bf a1 = load_A_tile(Xb, RSX, 16, ks * 32, lane);
      v16bf bt = load_B_tile(Winpk, ks * 32 + tn, lane);
      a0c = wmma_bf16(a0, bt, a0c);
      a1c = wmma_bf16(a1, bt, a1c);
    }
    int col = tn * 16 + (lane & 15);
    float bias = Win_b[col];
#pragma unroll
    for (int v = 0; v < 8; ++v) {
      int r0 = v + 8 * hl;
      int r1 = 16 + v + 8 * hl;
      Fb[r0 * RSF + col] = (bf16_t)gelu_erf_f(a0c[v] + bias);
      Fb[r1 * RSF + col] = (bf16_t)gelu_erf_f(a1c[v] + bias);
    }
  }
  __syncthreads();

  // dh2 = ff[32x512] @ Wout[512x128]; out = mask_V * (h + dh2 + b)
  v8f acc[2][2];
#pragma unroll
  for (int rt = 0; rt < 2; ++rt)
#pragma unroll
    for (int j = 0; j < 2; ++j) acc[rt][j] = vzero;
  for (int ks = 0; ks < 16; ++ks) {
    v16bf a0 = load_A_tile(Fb, RSF, 0, ks * 32, lane);
    v16bf a1 = load_A_tile(Fb, RSF, 16, ks * 32, lane);
#pragma unroll
    for (int j = 0; j < 2; ++j) {
      v16bf bt = load_B_tile(Woutpk, ks * 8 + (wave * 2 + j), lane);
      acc[0][j] = wmma_bf16(a0, bt, acc[0][j]);
      acc[1][j] = wmma_bf16(a1, bt, acc[1][j]);
    }
  }
#pragma unroll
  for (int rt = 0; rt < 2; ++rt)
#pragma unroll
    for (int j = 0; j < 2; ++j) {
      int col = (wave * 2 + j) * 16 + (lane & 15);
      float bias = Wout_b[col];
#pragma unroll
      for (int v = 0; v < 8; ++v) {
        int row = rt * 16 + v + 8 * hl;
        int g = base + row;
        float val = hbuf[(size_t)g * H_DIM + col] + acc[rt][j][v] + bias;
        out[(size_t)g * H_DIM + col] = mask_V[g] * val;
      }
    }
}

// ---- Host launcher ----------------------------------------------------------
extern "C" void kernel_launch(void* const* d_in, const int* in_sizes, int n_in,
                              void* d_out, int out_size, void* d_ws, size_t ws_size,
                              hipStream_t stream) {
  const float* h_V        = (const float*)d_in[0];
  const float* h_E        = (const float*)d_in[1];
  const int*   E_idx      = (const int*)d_in[2];
  const float* mask_V     = (const float*)d_in[3];
  const float* mask_att   = (const float*)d_in[4];
  const float* W1_w = (const float*)d_in[5];
  const float* W1_b = (const float*)d_in[6];
  const float* W2_w = (const float*)d_in[7];
  const float* W2_b = (const float*)d_in[8];
  const float* W3_w = (const float*)d_in[9];
  const float* W3_b = (const float*)d_in[10];
  const float* Win_w  = (const float*)d_in[11];
  const float* Win_b  = (const float*)d_in[12];
  const float* Wout_w = (const float*)d_in[13];
  const float* Wout_b = (const float*)d_in[14];
  float* out = (float*)d_out;

  char* ws = (char*)d_ws;
  size_t off = 0;
  auto alloc = [&](size_t bytes) {
    void* p = ws + off;
    off += (bytes + 255) & ~(size_t)255;
    return p;
  };
  bf16_t* W1pk   = (bf16_t*)alloc((size_t)C_IN * H_DIM * sizeof(bf16_t));
  bf16_t* W2pk   = (bf16_t*)alloc((size_t)H_DIM * H_DIM * sizeof(bf16_t));
  bf16_t* W3pk   = (bf16_t*)alloc((size_t)H_DIM * H_DIM * sizeof(bf16_t));
  bf16_t* Winpk  = (bf16_t*)alloc((size_t)H_DIM * FF_DIM * sizeof(bf16_t));
  bf16_t* Woutpk = (bf16_t*)alloc((size_t)FF_DIM * H_DIM * sizeof(bf16_t));
  float*  hbuf   = (float*)alloc((size_t)B_DIM * N_DIM * H_DIM * sizeof(float));

  pack_wmma_b<<<(C_IN * H_DIM + 255) / 256, 256, 0, stream>>>(W1_w, W1pk, C_IN, H_DIM);
  pack_wmma_b<<<(H_DIM * H_DIM + 255) / 256, 256, 0, stream>>>(W2_w, W2pk, H_DIM, H_DIM);
  pack_wmma_b<<<(H_DIM * H_DIM + 255) / 256, 256, 0, stream>>>(W3_w, W3pk, H_DIM, H_DIM);
  pack_wmma_b<<<(H_DIM * FF_DIM + 255) / 256, 256, 0, stream>>>(Win_w, Winpk, H_DIM, FF_DIM);
  pack_wmma_b<<<(FF_DIM * H_DIM + 255) / 256, 256, 0, stream>>>(Wout_w, Woutpk, FF_DIM, H_DIM);

  enc_msg_kernel<<<B_DIM * N_DIM, 128, 0, stream>>>(
      h_V, h_E, E_idx, mask_att, W1pk, W1_b, W2pk, W2_b, W3pk, W3_b, hbuf);

  enc_ffn_kernel<<<(B_DIM * N_DIM) / ROWS2, 128, 0, stream>>>(
      hbuf, mask_V, Winpk, Win_b, Woutpk, Wout_b, out);
}
